// UniversalReasoner_10342281249268
// MI455X (gfx1250) — compile-verified, matrix operand order checked by reading, not run on hardware
//
#include <hip/hip_runtime.h>
#include <hip/hip_bf16.h>

#define SCRATCH_N 16
#define NUM_BRANCHES 4
#define NUM_HEADS 8
#define MAX_STEPS 8
#define VOCAB 100277
#define DMODEL 1024
#define DFF 2048
#define BSZ 8
#define SEQ 128
#define TOTAL (SEQ + SCRATCH_N)   // 144
#define HD (DMODEL / NUM_HEADS)   // 128

typedef __attribute__((ext_vector_type(16))) _Float16 v16h;
typedef __attribute__((ext_vector_type(8)))  float    v8f;
typedef __attribute__((ext_vector_type(4)))  unsigned int v4u;
typedef __attribute__((ext_vector_type(8)))  unsigned int v8u;

__device__ __forceinline__ float gelu_tanh(float x) {
    float x3 = x * x * x;
    return 0.5f * x * (1.0f + tanhf(0.7978845608028654f * (x + 0.044715f * x3)));
}

// ---------------------------------------------------------------------------
// Issue a TDM load of a 64(X) x 32(Y) f32 tile of W (row stride Nn) into LDS.
// Descriptor per CDNA5 ISA. tensor_dim0 clamps the
// ragged N edge so OOB columns are zero-filled by the TDM itself.
// ---------------------------------------------------------------------------
__device__ __forceinline__ void tdm_issue_b(const float* W, int Kk, int Nn,
                                            int k0, int col0, unsigned int ldsAddr) {
    unsigned long long gaddr =
        (unsigned long long)(size_t)W + ((size_t)k0 * Nn + col0) * 4ull;
    unsigned int td0 = (unsigned int)(Nn - col0);
    unsigned int td1 = (unsigned int)(Kk - k0);
    v4u g0; v8u g1;
    g0[0] = 1u;                                   // count=1, user mode
    g0[1] = ldsAddr;                              // lds_addr (bytes)
    g0[2] = (unsigned int)gaddr;                  // global_addr[31:0]
    g0[3] = (unsigned int)((gaddr >> 32) & 0x01FFFFFFull) | (2u << 30); // type=2
    g1[0] = (2u << 16);                           // data_size=4B, wg_mask=0
    g1[1] = (td0 & 0xFFFFu) << 16;                // tensor_dim0[15:0]
    g1[2] = (td0 >> 16) | ((td1 & 0xFFFFu) << 16);// dim0[31:16] | dim1[15:0]
    g1[3] = (td1 >> 16) | (64u << 16);            // dim1[31:16] | tile_dim0=64
    g1[4] = 32u;                                  // tile_dim1=32, tile_dim2=0
    g1[5] = (unsigned int)Nn;                     // tensor_dim0_stride[31:0]
    g1[6] = 0u;
    g1[7] = 0u;
    asm volatile("tensor_load_to_lds %0, %1" :: "s"(g0), "s"(g1) : "memory");
}

// ---------------------------------------------------------------------------
// Gather: comb rows [b*144 + t]: t<128 -> embed[tokens[b,t]], else scratch row
// ---------------------------------------------------------------------------
__global__ void gather_kernel(const float* __restrict__ embed,
                              const float* __restrict__ scratch,
                              const int* __restrict__ tokens,
                              float* __restrict__ comb) {
    int r = blockIdx.x;                 // 0 .. B*TOTAL-1
    int b = r / TOTAL, t = r % TOTAL;
    const float* src;
    if (t < SEQ) src = embed + (size_t)tokens[b * SEQ + t] * DMODEL;
    else         src = scratch + (size_t)(t - SEQ) * DMODEL;
    float* dst = comb + (size_t)r * DMODEL;
    for (int d = threadIdx.x; d < DMODEL; d += 256) dst[d] = src[d];
}

// ---------------------------------------------------------------------------
// predicted_steps[b] = sigmoid(mean_s(z_seq[b] @ diffw) + diffb) * MAX_STEPS
// ---------------------------------------------------------------------------
__global__ void predsteps_kernel(const float* __restrict__ comb,
                                 const float* __restrict__ diffw,
                                 const float* __restrict__ diffb,
                                 float* __restrict__ out) {
    int b = blockIdx.x, tid = threadIdx.x;
    __shared__ float red[256];
    float sum = 0.f;
    for (int idx = tid; idx < SEQ * DMODEL; idx += 256) {
        int s = idx >> 10, d = idx & (DMODEL - 1);
        sum += comb[(size_t)(b * TOTAL + s) * DMODEL + d] * diffw[d];
    }
    red[tid] = sum; __syncthreads();
    for (int off = 128; off > 0; off >>= 1) {
        if (tid < off) red[tid] += red[tid + off];
        __syncthreads();
    }
    if (tid == 0) {
        float mean = red[0] / (float)SEQ + diffb[0];
        out[b] = (1.f / (1.f + __expf(-mean))) * (float)MAX_STEPS;
    }
}

// ---------------------------------------------------------------------------
// LayerNorm: one block per row
// ---------------------------------------------------------------------------
__global__ void ln_kernel(const float* __restrict__ X,
                          const float* __restrict__ sc,
                          const float* __restrict__ bi,
                          float* __restrict__ Y) {
    int r = blockIdx.x, tid = threadIdx.x;
    const float* x = X + (size_t)r * DMODEL;
    __shared__ float s1[256], s2[256];
    float a = 0.f, b = 0.f;
    for (int i = tid; i < DMODEL; i += 256) { float v = x[i]; a += v; b += v * v; }
    s1[tid] = a; s2[tid] = b; __syncthreads();
    for (int off = 128; off > 0; off >>= 1) {
        if (tid < off) { s1[tid] += s1[tid + off]; s2[tid] += s2[tid + off]; }
        __syncthreads();
    }
    float mean = s1[0] * (1.f / DMODEL);
    float var  = s2[0] * (1.f / DMODEL) - mean * mean;
    float rstd = rsqrtf(var + 1e-6f);
    float* y = Y + (size_t)r * DMODEL;
    for (int i = tid; i < DMODEL; i += 256)
        y[i] = (x[i] - mean) * rstd * sc[i] + bi[i];
}

// ---------------------------------------------------------------------------
// WMMA GEMM, TDM-pipelined.
//   Block = 256 threads (8 waves). Block tile: 128(M) x 64(N), BK = 32.
//   - B tiles are DMA'd by the Tensor Data Mover into a double-buffered LDS
//     staging area; wave 0 issues the next tile then s_wait_tensorcnt(1) for
//     the in-flight one (TDM ops complete in order within a wave).
//   - One cooperative pass converts the f32 tile into transposed f16 Bt16 so
//     each B fragment is two contiguous ds_load_b128's and every element is
//     converted exactly once per block.
//   - A fragments load straight from global (each element touched once per
//     block; cross-block N reuse rides the 192 MB L2), with prefetch.
//   - Wave w owns rows [w*16,w*16+16) x 64 cols: 4 WMMAs, A-fragment reuse.
// ---------------------------------------------------------------------------
__global__ void __launch_bounds__(256)
gemm_wmma(const float* __restrict__ A, const float* __restrict__ W,
          const float* __restrict__ bias, const float* __restrict__ resid,
          float* __restrict__ C, int Kk, int Nn, int act) {
    __shared__ float    Bt32[2][32 * 64];  // 2 x 8 KB TDM destination
    __shared__ _Float16 Bt16[64 * 32];     // 4 KB transposed f16 tile

    const int tid  = threadIdx.x;
    const int lane = tid & 31;
    const int wv   = tid >> 5;
    const int col0 = blockIdx.x * 64;
    const int row0 = blockIdx.y * 128;
    const int kb   = (lane >> 4) << 3;     // 0 or 8
    const int mrow = row0 + (wv << 4) + (lane & 15);
    const float* arow = A + (size_t)mrow * Kk;

    union { v16h v; _Float16 e[16]; } af, bf;
    union { v8f  v; float    e[8];  } acc[4];
#pragma unroll
    for (int nt = 0; nt < 4; ++nt)
#pragma unroll
        for (int i = 0; i < 8; ++i) acc[nt].e[i] = 0.f;

    if (wv == 0)
        tdm_issue_b(W, Kk, Nn, 0, col0, (unsigned int)(size_t)&Bt32[0][0]);

    for (int k0 = 0; k0 < Kk; k0 += 32) {
        const int ib = (k0 >> 5) & 1;

        if (wv == 0) {
            if (k0 + 32 < Kk) {
                tdm_issue_b(W, Kk, Nn, k0 + 32, col0,
                            (unsigned int)(size_t)&Bt32[ib ^ 1][0]);
                __builtin_amdgcn_s_wait_tensorcnt(1);  // in-order: tile ib landed
            } else {
                __builtin_amdgcn_s_wait_tensorcnt(0);
            }
        }

        // A fragment straight from global: 2x contiguous 8-float chunks
        if (k0 + 32 < Kk) __builtin_prefetch(arow + k0 + 32, 0, 1);
#pragma unroll
        for (int i = 0; i < 8; ++i) af.e[i]     = (_Float16)arow[k0 + kb + i];
#pragma unroll
        for (int i = 0; i < 8; ++i) af.e[8 + i] = (_Float16)arow[k0 + 16 + kb + i];

        __syncthreads();   // Bt32[ib] complete; previous Bt16 reads consumed

        // cooperative transpose+convert: each element converted once per block
#pragma unroll
        for (int i = tid; i < 32 * 64; i += 256) {
            int k = i >> 6, n = i & 63;
            Bt16[n * 32 + k] = (_Float16)Bt32[ib][k * 64 + n];
        }
        __syncthreads();   // Bt16 ready

#pragma unroll
        for (int nt = 0; nt < 4; ++nt) {
            int nloc = (nt << 4) + (lane & 15);
#pragma unroll
            for (int i = 0; i < 8; ++i) bf.e[i]     = Bt16[nloc * 32 + kb + i];
#pragma unroll
            for (int i = 0; i < 8; ++i) bf.e[8 + i] = Bt16[nloc * 32 + 16 + kb + i];
            acc[nt].v = __builtin_amdgcn_wmma_f32_16x16x32_f16(
                false, af.v, false, bf.v, (short)0, acc[nt].v, false, false);
        }
    }

    // ---- epilogue: bias / residual / activation / store (N-guarded) ----
#pragma unroll
    for (int nt = 0; nt < 4; ++nt) {
        int ncol = col0 + (nt << 4) + (lane & 15);
        if (ncol >= Nn) continue;
        float bv = bias ? bias[ncol] : 0.f;
#pragma unroll
        for (int r = 0; r < 8; ++r) {
            int m = row0 + (wv << 4) + r + ((lane >> 4) << 3);
            float v = acc[nt].e[r] + bv;
            if (resid) v += resid[(size_t)m * Nn + ncol];
            if (act == 1) v = gelu_tanh(v);
            C[(size_t)m * Nn + ncol] = v;
        }
    }
}

// ---------------------------------------------------------------------------
// RoPE in place on (B*TOTAL, D) viewed as (b,t,H,HD); pos = t
// ---------------------------------------------------------------------------
__global__ void rope_kernel(float* __restrict__ X) {
    int row = blockIdx.x;               // 0 .. B*TOTAL-1
    int t = row % TOTAL;
    int p = blockIdx.y * 256 + threadIdx.x;   // 0..511 pair index
    if (p >= (DMODEL / 2)) return;
    int h = p >> 6, i = p & 63;
    float invf = __expf(-((2.f * (float)i) / (float)HD) * 9.210340371976184f); // ln(1e4)
    float ang = (float)t * invf;
    float c = __cosf(ang), s = __sinf(ang);
    size_t base = (size_t)row * DMODEL + h * HD + 2 * i;
    float xe = X[base], xo = X[base + 1];
    X[base]     = xe * c - xo * s;
    X[base + 1] = xo * c + xe * s;
}

// ---------------------------------------------------------------------------
// Attention: one block per (b,h,tq); keys/values over 144 rows of length 128
// ---------------------------------------------------------------------------
__global__ void attn_kernel(const float* __restrict__ Q, const float* __restrict__ K,
                            const float* __restrict__ V, float* __restrict__ O) {
    int idx = blockIdx.x;
    int tq = idx % TOTAL;
    int h  = (idx / TOTAL) % NUM_HEADS;
    int b  = idx / (TOTAL * NUM_HEADS);
    int tid = threadIdx.x;
    __shared__ float qsh[HD];
    __shared__ float lg[TOTAL];
    __shared__ float red[256];

    const float* qrow = Q + (size_t)(b * TOTAL + tq) * DMODEL + h * HD;
    if (tid < HD) qsh[tid] = qrow[tid];
    __syncthreads();

    if (tid < TOTAL) {
        const float* krow = K + (size_t)(b * TOTAL + tid) * DMODEL + h * HD;
        float s = 0.f;
        for (int d = 0; d < HD; ++d) s += qsh[d] * krow[d];
        bool masked = (tq < SEQ) && (tid > tq);   // scratch queries see all keys
        lg[tid] = masked ? -1e9f : s * 0.08838834764831845f;  // 1/sqrt(128)
    }
    __syncthreads();

    float m = -1e30f;
    if (tid < TOTAL) m = lg[tid];
    red[tid] = m; __syncthreads();
    for (int off = 128; off > 0; off >>= 1) {
        if (tid < off) red[tid] = fmaxf(red[tid], red[tid + off]);
        __syncthreads();
    }
    m = red[0]; __syncthreads();

    float e = 0.f;
    if (tid < TOTAL) { e = __expf(lg[tid] - m); lg[tid] = e; }
    red[tid] = e; __syncthreads();
    for (int off = 128; off > 0; off >>= 1) {
        if (tid < off) red[tid] += red[tid + off];
        __syncthreads();
    }
    float inv = 1.f / red[0];
    __syncthreads();

    if (tid < HD) {
        float o = 0.f;
        const float* vb = V + (size_t)(b * TOTAL) * DMODEL + h * HD + tid;
        for (int j = 0; j < TOTAL; ++j) o += lg[j] * vb[(size_t)j * DMODEL];
        O[(size_t)(b * TOTAL + tq) * DMODEL + h * HD + tid] = o * inv;
    }
}

// ---------------------------------------------------------------------------
// Halting: halt[b] = sigmoid(mean over 16 scratch rows of (row . hw) + hb)
// ---------------------------------------------------------------------------
__global__ void halt_kernel(const float* __restrict__ xout,
                            const float* __restrict__ hw, const float* __restrict__ hb,
                            float* __restrict__ cum, float* __restrict__ scale,
                            float* __restrict__ halts, int step, int isLast) {
    int b = blockIdx.x, tid = threadIdx.x;
    __shared__ float red[256];
    float sum = 0.f;
    for (int idx = tid; idx < SCRATCH_N * DMODEL; idx += 256) {
        int j = idx >> 10, d = idx & (DMODEL - 1);
        sum += xout[(size_t)(b * TOTAL + SEQ + j) * DMODEL + d] * hw[d];
    }
    red[tid] = sum; __syncthreads();
    for (int off = 128; off > 0; off >>= 1) {
        if (tid < off) red[tid] += red[tid + off];
        __syncthreads();
    }
    if (tid == 0) {
        float mean = red[0] / (float)SCRATCH_N + hb[0];
        float halt = 1.f / (1.f + __expf(-mean));
        float c = cum[b];
        float p = halt * (1.f - c);
        float cnew = c + p;
        cum[b] = cnew;
        float sc = p, oh = p;
        if (isLast) { float rem = 1.f - cnew; sc = p + rem; oh = p + rem; }
        scale[b] = sc;
        halts[b * MAX_STEPS + step] = oh;
    }
}

// comb rows 128..143 <- xout rows 128..143 (z_s update; branch mixer == identity)
__global__ void copy_zs_kernel(const float* __restrict__ xout, float* __restrict__ comb) {
    int r = blockIdx.x;                 // 0..B*16-1
    int b = r >> 4, j = r & 15;
    size_t row = (size_t)(b * TOTAL + SEQ + j) * DMODEL;
    for (int d = threadIdx.x; d < DMODEL; d += 256) comb[row + d] = xout[row + d];
}

// acc[b,s,:] += fused[b*144+s,:] * scale[b]
__global__ void acc_kernel(const float* __restrict__ fused,
                           const float* __restrict__ scale,
                           float* __restrict__ acc) {
    int r = blockIdx.x;                 // 0..B*S-1
    int b = r >> 7;
    int s = r & (SEQ - 1);
    float sc = scale[b];
    size_t src = (size_t)(b * TOTAL + s) * DMODEL;
    size_t dst = (size_t)r * DMODEL;
    for (int d = threadIdx.x; d < DMODEL; d += 256) acc[dst + d] += fused[src + d] * sc;
}

__global__ void zero_kernel(float* __restrict__ p, size_t n) {
    size_t i = (size_t)blockIdx.x * 256 + threadIdx.x;
    size_t stride = (size_t)gridDim.x * 256;
    for (; i < n; i += stride) p[i] = 0.f;
}

// ---------------------------------------------------------------------------
// Host orchestration
// ---------------------------------------------------------------------------
static inline void launch_gemm(const float* A, const float* W, const float* bias,
                               const float* resid, float* C,
                               int R, int K, int N, int act, hipStream_t st) {
    dim3 grid((N + 63) / 64, R / 128);
    gemm_wmma<<<grid, 256, 0, st>>>(A, W, bias, resid, C, K, N, act);
}

extern "C" void kernel_launch(void* const* d_in, const int* in_sizes, int n_in,
                              void* d_out, int out_size, void* d_ws, size_t ws_size,
                              hipStream_t stream) {
    const float* embed  = (const float*)d_in[0];
    const float* scr    = (const float*)d_in[1];
    const float* qw = (const float*)d_in[2];  const float* qb = (const float*)d_in[3];
    const float* kw = (const float*)d_in[4];  const float* kb = (const float*)d_in[5];
    const float* vw = (const float*)d_in[6];  const float* vb = (const float*)d_in[7];
    const float* ow = (const float*)d_in[8];  const float* ob = (const float*)d_in[9];
    const float* n1s = (const float*)d_in[10]; const float* n1b = (const float*)d_in[11];
    const float* f1w = (const float*)d_in[12]; const float* f1b = (const float*)d_in[13];
    const float* f2w = (const float*)d_in[14]; const float* f2b = (const float*)d_in[15];
    const float* n2s = (const float*)d_in[16]; const float* n2b = (const float*)d_in[17];
    // bsw/bsb (18,19) unused: branch mixer is mathematically the identity
    const float* hw = (const float*)d_in[20]; const float* hb = (const float*)d_in[21];
    const float* diffw = (const float*)d_in[22]; const float* diffb = (const float*)d_in[23];
    const float* decw = (const float*)d_in[24]; const float* decb = (const float*)d_in[25];
    const int* tokens = (const int*)d_in[26];

    const int R = BSZ * TOTAL;          // 1152 = 9 * 128
    const size_t RD = (size_t)R * DMODEL;

    float* ws      = (float*)d_ws;
    float* comb    = ws;                 // R x D (rows 0..127 per b = z_seq, 128..143 = z_s)
    float* lnbuf   = comb + RD;
    float* Qb      = lnbuf + RD;
    float* Kb      = Qb + RD;
    float* Vb      = Kb + RD;
    float* attnout = Vb + RD;
    float* xmid    = attnout + RD;       // also "fused" in readout
    float* xout    = xmid + RD;          // also "fusedout" in readout
    float* h1      = xout + RD;          // R x DFF
    float* accbuf  = h1 + (size_t)R * DFF;            // B*S x D
    float* cum     = accbuf + (size_t)BSZ * SEQ * DMODEL;
    float* scale   = cum + BSZ;

    float* out_logits = (float*)d_out;
    float* out_halts  = out_logits + (size_t)BSZ * SEQ * VOCAB;
    float* out_preds  = out_halts + BSZ * MAX_STEPS;

    // init
    zero_kernel<<<1024, 256, 0, stream>>>(accbuf, (size_t)BSZ * SEQ * DMODEL);
    zero_kernel<<<1, 256, 0, stream>>>(cum, (size_t)BSZ);

    // embed gather + scratch init into comb
    gather_kernel<<<R, 256, 0, stream>>>(embed, scr, tokens, comb);
    predsteps_kernel<<<BSZ, 256, 0, stream>>>(comb, diffw, diffb, out_preds);

    dim3 rope_grid(R, 2);
    const int attn_blocks = BSZ * NUM_HEADS * TOTAL;

    for (int step = 0; step < MAX_STEPS; ++step) {
        int isLast = (step == MAX_STEPS - 1) ? 1 : 0;

        // ---- processor: x_mid = comb + attn(ln1(comb)); x_out = x_mid + ffn(ln2(x_mid))
        ln_kernel<<<R, 256, 0, stream>>>(comb, n1s, n1b, lnbuf);
        launch_gemm(lnbuf, qw, qb, nullptr, Qb, R, DMODEL, DMODEL, 0, stream);
        launch_gemm(lnbuf, kw, kb, nullptr, Kb, R, DMODEL, DMODEL, 0, stream);
        launch_gemm(lnbuf, vw, vb, nullptr, Vb, R, DMODEL, DMODEL, 0, stream);
        rope_kernel<<<rope_grid, 256, 0, stream>>>(Qb);
        rope_kernel<<<rope_grid, 256, 0, stream>>>(Kb);
        attn_kernel<<<attn_blocks, 256, 0, stream>>>(Qb, Kb, Vb, attnout);
        launch_gemm(attnout, ow, ob, comb, xmid, R, DMODEL, DMODEL, 0, stream);
        ln_kernel<<<R, 256, 0, stream>>>(xmid, n2s, n2b, lnbuf);
        launch_gemm(lnbuf, f1w, f1b, nullptr, h1, R, DMODEL, DFF, 1, stream);
        launch_gemm(h1, f2w, f2b, xmid, xout, R, DFF, DMODEL, 0, stream);

        // ---- halting on z_s_new = xout[:, -16:, :]
        halt_kernel<<<BSZ, 256, 0, stream>>>(xout, hw, hb, cum, scale,
                                             out_halts, step, isLast);
        // ---- update z_s rows of comb
        copy_zs_kernel<<<BSZ * SCRATCH_N, 256, 0, stream>>>(xout, comb);

        // ---- readout: fused = attn(ln1(comb)); fused += ffn(ln2(fused)); acc += fused[:, :S]*scale
        ln_kernel<<<R, 256, 0, stream>>>(comb, n1s, n1b, lnbuf);
        launch_gemm(lnbuf, qw, qb, nullptr, Qb, R, DMODEL, DMODEL, 0, stream);
        launch_gemm(lnbuf, kw, kb, nullptr, Kb, R, DMODEL, DMODEL, 0, stream);
        launch_gemm(lnbuf, vw, vb, nullptr, Vb, R, DMODEL, DMODEL, 0, stream);
        rope_kernel<<<rope_grid, 256, 0, stream>>>(Qb);
        rope_kernel<<<rope_grid, 256, 0, stream>>>(Kb);
        attn_kernel<<<attn_blocks, 256, 0, stream>>>(Qb, Kb, Vb, attnout);
        launch_gemm(attnout, ow, ob, nullptr, xmid, R, DMODEL, DMODEL, 0, stream);  // fused
        ln_kernel<<<R, 256, 0, stream>>>(xmid, n2s, n2b, lnbuf);
        launch_gemm(lnbuf, f1w, f1b, nullptr, h1, R, DMODEL, DFF, 1, stream);
        launch_gemm(h1, f2w, f2b, xmid, xout, R, DFF, DMODEL, 0, stream);           // fusedout
        acc_kernel<<<BSZ * SEQ, 256, 0, stream>>>(xout, scale, accbuf);
    }

    // ---- decode: logits = final_latent @ decw + decb  (bandwidth-bound on decw)
    launch_gemm(accbuf, decw, decb, nullptr, out_logits,
                BSZ * SEQ, DMODEL, VOCAB, 0, stream);
}
